// MoeLayer_8890582303068
// MI455X (gfx1250) — matrix operand order repaired and checked by reading, not compiled, not run on hardware
//
#include <hip/hip_runtime.h>
#include <hip/hip_bf16.h>
#include <math.h>

// ---------------- problem constants ----------------
#define NEXP    8
#define TOKENS  8192          // B*S = 4*2048
#define DIM     1024          // D
#define DFF     4096          // DF
#define NASSIGN (2*TOKENS)    // top-2: exactly 2 assignments per token

// ---------------- GEMM tiling ----------------
#define TM 128
#define TN 128
#define TK 32
#define LDSS 40               // LDS row stride in bf16 elements (80B, 16B aligned, depitched)

typedef __attribute__((ext_vector_type(16))) __bf16    v16bf;
typedef __attribute__((ext_vector_type(8)))  float     v8f;
typedef __attribute__((ext_vector_type(4)))  unsigned  u32x4;
typedef __attribute__((ext_vector_type(2)))  unsigned  u32x2;

union FragBF { u32x4 q[2]; v16bf v; };

__device__ __forceinline__ unsigned short f2bf(float f) {
    unsigned u = __float_as_uint(f);
    u += 0x7FFFu + ((u >> 16) & 1u);            // round-to-nearest-even
    return (unsigned short)(u >> 16);
}

__device__ __forceinline__ float gelu_tanh(float x) {
    const float c = 0.7978845608028654f;        // sqrt(2/pi)
    float t = tanhf(c * (x + 0.044715f * x * x * x));
    return 0.5f * x * (1.0f + t);
}

__device__ __forceinline__ void atomic_add_f32(float* p, float v) {
    __hip_atomic_fetch_add(p, v, __ATOMIC_RELAXED, __HIP_MEMORY_SCOPE_AGENT);
}

// CDNA5 direct global->LDS async copy (ASYNCcnt path, no VGPR staging).
// lds = LDS byte offset (low 32 bits of generic pointer to __shared__), g = per-lane global address.
__device__ __forceinline__ void async_ld128(unsigned lds, const void* g) {
    asm volatile("global_load_async_to_lds_b128 %0, %1, off" :: "v"(lds), "v"(g) : "memory");
}
// Same, but fetches the NEXT K-slab: displacement folded into the immediate offset (TK*2 = 64B).
__device__ __forceinline__ void async_ld128_next(unsigned lds, const void* g) {
    asm volatile("global_load_async_to_lds_b128 %0, %1, off offset:64" :: "v"(lds), "v"(g) : "memory");
}
__device__ __forceinline__ void wait_async0() {
    asm volatile("s_wait_asynccnt 0" ::: "memory");
}
__device__ __forceinline__ unsigned lds_off(const void* p) {
    return (unsigned)(uintptr_t)p;   // LDS aperture: addr[31:0] is the LDS byte offset
}

// ---------------- x: f32 -> bf16 ----------------
__global__ void k_convert_x(const float* __restrict__ x, unsigned short* __restrict__ xb) {
    size_t i = ((size_t)blockIdx.x * blockDim.x + threadIdx.x) * 4;
    float4 v = *(const float4*)(x + i);
    u32x2 o;
    o.x = (unsigned)f2bf(v.x) | ((unsigned)f2bf(v.y) << 16);
    o.y = (unsigned)f2bf(v.z) | ((unsigned)f2bf(v.w) << 16);
    *(u32x2*)(xb + i) = o;
}

// ---------------- per-expert weight transpose: f32 [R][C] -> bf16 [C][R] ----------------
__global__ void k_transpose_bf16(const float* __restrict__ src, unsigned short* __restrict__ dst,
                                 int R, int C) {
    __shared__ unsigned short tile[32][33];
    const float* s = src + (size_t)blockIdx.z * R * C;
    unsigned short* d = dst + (size_t)blockIdx.z * R * C;
    int c0 = blockIdx.x * 32, r0 = blockIdx.y * 32;
    int tx = threadIdx.x, ty = threadIdx.y;
#pragma unroll
    for (int i = 0; i < 4; ++i)
        tile[ty + 8 * i][tx] = f2bf(s[(size_t)(r0 + ty + 8 * i) * C + c0 + tx]);
    __syncthreads();
#pragma unroll
    for (int i = 0; i < 4; ++i)
        d[(size_t)(c0 + ty + 8 * i) * R + r0 + tx] = tile[tx][ty + 8 * i];
}

// ---------------- gating: logits, top-2 softmax, expert counts ----------------
__global__ void k_gate(const float* __restrict__ x, const float* __restrict__ Wg,
                       int* __restrict__ cnt, int2* __restrict__ tok_e, float2* __restrict__ tok_p) {
    int t = blockIdx.x * 8 + (threadIdx.x >> 5);   // one wave32 per token
    int lane = threadIdx.x & 31;
    float acc[NEXP];
#pragma unroll
    for (int e = 0; e < NEXP; ++e) acc[e] = 0.0f;
    const float* xr = x + (size_t)t * DIM;
    for (int d = lane; d < DIM; d += 32) {
        float xv = xr[d];
        const float* wr = Wg + (size_t)d * NEXP;
#pragma unroll
        for (int e = 0; e < NEXP; ++e) acc[e] += xv * wr[e];
    }
#pragma unroll
    for (int off = 16; off > 0; off >>= 1) {
#pragma unroll
        for (int e = 0; e < NEXP; ++e) acc[e] += __shfl_down(acc[e], off, 32);
    }
    if (lane == 0) {
        int b0 = 0;
#pragma unroll
        for (int e = 1; e < NEXP; ++e) if (acc[e] > acc[b0]) b0 = e;
        int b1 = (b0 == 0) ? 1 : 0;
#pragma unroll
        for (int e = 0; e < NEXP; ++e) if (e != b0 && acc[e] > acc[b1]) b1 = e;
        float z = __expf(acc[b1] - acc[b0]);       // acc[b1] <= acc[b0]
        float p0 = 1.0f / (1.0f + z);
        float p1 = 1.0f - p0;
        atomicAdd(&cnt[b0], 1);
        atomicAdd(&cnt[b1], 1);
        tok_e[t] = make_int2(b0, b1);
        tok_p[t] = make_float2(p0, p1);
    }
}

// ---------------- tiny prefix sum over 8 experts ----------------
__global__ void k_prefix(const int* __restrict__ cnt, int* __restrict__ base, int* __restrict__ cnt2) {
    if (threadIdx.x == 0) {
        int s = 0;
#pragma unroll
        for (int e = 0; e < NEXP; ++e) { base[e] = s; s += cnt[e]; }
    }
    if (threadIdx.x < NEXP) cnt2[threadIdx.x] = 0;
}

// ---------------- scatter tokens into expert-grouped assignment rows ----------------
__global__ void k_scatter(const int2* __restrict__ tok_e, const float2* __restrict__ tok_p,
                          const int* __restrict__ base, int* __restrict__ cnt2,
                          int* __restrict__ rows, float* __restrict__ wts) {
    int t = blockIdx.x * blockDim.x + threadIdx.x;
    if (t >= TOKENS) return;
    int2 ee = tok_e[t];
    float2 pp = tok_p[t];
    int s0 = atomicAdd(&cnt2[ee.x], 1); int r0 = base[ee.x] + s0; rows[r0] = t; wts[r0] = pp.x;
    int s1 = atomicAdd(&cnt2[ee.y], 1); int r1 = base[ee.y] + s1; rows[r1] = t; wts[r1] = pp.y;
}

// ---------------- GEMM1: H = gelu(X[rows] @ W1^T + b1), bf16 out ----------------
// A: xb [TOKENS][DIM] bf16 (gathered rows); B: w1t [E][DFF][DIM] bf16 (N-major, K-contig)
__global__ __launch_bounds__(256) void k_ffn1(
    const unsigned short* __restrict__ xb, const unsigned short* __restrict__ w1t,
    const float* __restrict__ b1, const int* __restrict__ rows,
    const int* __restrict__ base, const int* __restrict__ cnt,
    unsigned short* __restrict__ H) {
    const int e = blockIdx.z;
    const int rbeg = base[e];
    const int rend = rbeg + cnt[e];
    const int m0 = rbeg + blockIdx.y * TM;
    if (m0 >= rend) return;
    const int n0 = blockIdx.x * TN;

    __shared__ __align__(16) unsigned short As[2][TM * LDSS];
    __shared__ __align__(16) unsigned short Bs[2][TN * LDSS];

    const int tid = threadIdx.x;
    const int lane = tid & 31;
    const int wave = tid >> 5;
    const int wm = wave & 3;    // 4 waves along M (32 rows each)
    const int wn = wave >> 2;   // 2 waves along N (64 cols each)

    const int lr = tid >> 2;          // loader row 0..63 (two passes)
    const int lc = (tid & 3) * 8;     // element offset {0,8,16,24}

    int tok0 = rows[min(m0 + lr, rend - 1)];
    int tok1 = rows[min(m0 + lr + 64, rend - 1)];
    const unsigned short* pa0 = xb + (size_t)tok0 * DIM + lc;       // advancing pointers
    const unsigned short* pa1 = xb + (size_t)tok1 * DIM + lc;
    const unsigned short* bp  = w1t + (size_t)e * DFF * DIM + (size_t)n0 * DIM;
    const unsigned short* pb0 = bp + (size_t)lr * DIM + lc;
    const unsigned short* pb1 = bp + (size_t)(lr + 64) * DIM + lc;

    unsigned sA0[2], sA1[2], sB0[2], sB1[2];
#pragma unroll
    for (int b = 0; b < 2; ++b) {
        sA0[b] = lds_off(&As[b][lr * LDSS + lc]);
        sA1[b] = lds_off(&As[b][(lr + 64) * LDSS + lc]);
        sB0[b] = lds_off(&Bs[b][lr * LDSS + lc]);
        sB1[b] = lds_off(&Bs[b][(lr + 64) * LDSS + lc]);
    }

    v8f acc[2][4];
    const v8f vz = {0.f, 0.f, 0.f, 0.f, 0.f, 0.f, 0.f, 0.f};
#pragma unroll
    for (int mi = 0; mi < 2; ++mi)
#pragma unroll
        for (int ni = 0; ni < 4; ++ni) acc[mi][ni] = vz;

    // preload tile 0 (direct global->LDS, ASYNCcnt)
    async_ld128(sA0[0], pa0);
    async_ld128(sA1[0], pa1);
    async_ld128(sB0[0], pb0);
    async_ld128(sB1[0], pb1);
    wait_async0();
    __syncthreads();

    int buf = 0;
    for (int k0 = 0; k0 < DIM; k0 += TK) {
        if (k0 + TK < DIM) {   // prefetch next K-slab (p + 64B via immediate offset)
            async_ld128_next(sA0[buf ^ 1], pa0);
            async_ld128_next(sA1[buf ^ 1], pa1);
            async_ld128_next(sB0[buf ^ 1], pb0);
            async_ld128_next(sB1[buf ^ 1], pb1);
        }

        const unsigned short* Ab = As[buf];
        const unsigned short* Bb = Bs[buf];
        FragBF afr[2], bfr[4];
        const int hrow = lane & 15;
        const int ak = (lane < 16) ? 0 : 8;   // A: lane<16 -> K{0..7,16..23}; lane>=16 -> K{8..15,24..31}
        const int bk = (lane < 16) ? 0 : 8;   // B: lane<16 -> K{0..7,16..23} of col N; lane>=16 -> K{8..15,24..31}
#pragma unroll
        for (int mi = 0; mi < 2; ++mi) {
            int r = (wm * 32 + mi * 16 + hrow) * LDSS;
            afr[mi].q[0] = *(const u32x4*)&Ab[r + ak];
            afr[mi].q[1] = *(const u32x4*)&Ab[r + ak + 16];
        }
#pragma unroll
        for (int ni = 0; ni < 4; ++ni) {
            int r = (wn * 64 + ni * 16 + hrow) * LDSS;
            bfr[ni].q[0] = *(const u32x4*)&Bb[r + bk];
            bfr[ni].q[1] = *(const u32x4*)&Bb[r + bk + 16];
        }
#pragma unroll
        for (int mi = 0; mi < 2; ++mi)
#pragma unroll
            for (int ni = 0; ni < 4; ++ni)
                acc[mi][ni] = __builtin_amdgcn_wmma_f32_16x16x32_bf16(
                    false, afr[mi].v, false, bfr[ni].v, (short)0, acc[mi][ni], false, false);

        pa0 += TK; pa1 += TK; pb0 += TK; pb1 += TK;

        wait_async0();        // our prefetch landed
        __syncthreads();      // everyone's prefetch landed / reads of old buf done
        buf ^= 1;
    }

    const int mlane = (lane < 16) ? 0 : 8;
#pragma unroll
    for (int ni = 0; ni < 4; ++ni) {
        int gn = n0 + wn * 64 + ni * 16 + (lane & 15);
        float bv = b1[(size_t)e * DFF + gn];
#pragma unroll
        for (int mi = 0; mi < 2; ++mi) {
#pragma unroll
            for (int r = 0; r < 8; ++r) {
                int arow = m0 + wm * 32 + mi * 16 + mlane + r;
                if (arow < rend) {
                    float v = gelu_tanh(acc[mi][ni][r] + bv);
                    H[(size_t)arow * DFF + gn] = f2bf(v);
                }
            }
        }
    }
}

// ---------------- GEMM2: out[token] += w * (H @ W2^T + b2), f32 atomic accumulate ----------------
// A: H [NASSIGN][DFF] bf16; B: w2t [E][DIM][DFF] bf16 (N-major, K-contig)
__global__ __launch_bounds__(256) void k_ffn2(
    const unsigned short* __restrict__ H, const unsigned short* __restrict__ w2t,
    const float* __restrict__ b2, const int* __restrict__ rows, const float* __restrict__ wts,
    const int* __restrict__ base, const int* __restrict__ cnt,
    float* __restrict__ out) {
    const int e = blockIdx.z;
    const int rbeg = base[e];
    const int rend = rbeg + cnt[e];
    const int m0 = rbeg + blockIdx.y * TM;
    if (m0 >= rend) return;
    const int n0 = blockIdx.x * TN;

    __shared__ __align__(16) unsigned short As[2][TM * LDSS];
    __shared__ __align__(16) unsigned short Bs[2][TN * LDSS];

    const int tid = threadIdx.x;
    const int lane = tid & 31;
    const int wave = tid >> 5;
    const int wm = wave & 3;
    const int wn = wave >> 2;

    const int lr = tid >> 2;
    const int lc = (tid & 3) * 8;

    const unsigned short* pa0 = H + (size_t)min(m0 + lr, rend - 1) * DFF + lc;
    const unsigned short* pa1 = H + (size_t)min(m0 + lr + 64, rend - 1) * DFF + lc;
    const unsigned short* bp  = w2t + (size_t)e * DIM * DFF + (size_t)n0 * DFF;
    const unsigned short* pb0 = bp + (size_t)lr * DFF + lc;
    const unsigned short* pb1 = bp + (size_t)(lr + 64) * DFF + lc;

    unsigned sA0[2], sA1[2], sB0[2], sB1[2];
#pragma unroll
    for (int b = 0; b < 2; ++b) {
        sA0[b] = lds_off(&As[b][lr * LDSS + lc]);
        sA1[b] = lds_off(&As[b][(lr + 64) * LDSS + lc]);
        sB0[b] = lds_off(&Bs[b][lr * LDSS + lc]);
        sB1[b] = lds_off(&Bs[b][(lr + 64) * LDSS + lc]);
    }

    v8f acc[2][4];
    const v8f vz = {0.f, 0.f, 0.f, 0.f, 0.f, 0.f, 0.f, 0.f};
#pragma unroll
    for (int mi = 0; mi < 2; ++mi)
#pragma unroll
        for (int ni = 0; ni < 4; ++ni) acc[mi][ni] = vz;

    async_ld128(sA0[0], pa0);
    async_ld128(sA1[0], pa1);
    async_ld128(sB0[0], pb0);
    async_ld128(sB1[0], pb1);
    wait_async0();
    __syncthreads();

    int buf = 0;
    for (int k0 = 0; k0 < DFF; k0 += TK) {
        if (k0 + TK < DFF) {
            async_ld128_next(sA0[buf ^ 1], pa0);
            async_ld128_next(sA1[buf ^ 1], pa1);
            async_ld128_next(sB0[buf ^ 1], pb0);
            async_ld128_next(sB1[buf ^ 1], pb1);
        }

        const unsigned short* Ab = As[buf];
        const unsigned short* Bb = Bs[buf];
        FragBF afr[2], bfr[4];
        const int hrow = lane & 15;
        const int ak = (lane < 16) ? 0 : 8;
        const int bk = (lane < 16) ? 0 : 8;
#pragma unroll
        for (int mi = 0; mi < 2; ++mi) {
            int r = (wm * 32 + mi * 16 + hrow) * LDSS;
            afr[mi].q[0] = *(const u32x4*)&Ab[r + ak];
            afr[mi].q[1] = *(const u32x4*)&Ab[r + ak + 16];
        }
#pragma unroll
        for (int ni = 0; ni < 4; ++ni) {
            int r = (wn * 64 + ni * 16 + hrow) * LDSS;
            bfr[ni].q[0] = *(const u32x4*)&Bb[r + bk];
            bfr[ni].q[1] = *(const u32x4*)&Bb[r + bk + 16];
        }
#pragma unroll
        for (int mi = 0; mi < 2; ++mi)
#pragma unroll
            for (int ni = 0; ni < 4; ++ni)
                acc[mi][ni] = __builtin_amdgcn_wmma_f32_16x16x32_bf16(
                    false, afr[mi].v, false, bfr[ni].v, (short)0, acc[mi][ni], false, false);

        pa0 += TK; pa1 += TK; pb0 += TK; pb1 += TK;

        wait_async0();
        __syncthreads();
        buf ^= 1;
    }

    const int mlane = (lane < 16) ? 0 : 8;
#pragma unroll
    for (int ni = 0; ni < 4; ++ni) {
        int gn = n0 + wn * 64 + ni * 16 + (lane & 15);
        float bv = b2[(size_t)e * DIM + gn];
#pragma unroll
        for (int mi = 0; mi < 2; ++mi) {
#pragma unroll
            for (int r = 0; r < 8; ++r) {
                int arow = m0 + wm * 32 + mi * 16 + mlane + r;
                if (arow < rend) {
                    float w = wts[arow];
                    int token = rows[arow];
                    atomic_add_f32(&out[(size_t)token * DIM + gn], w * (acc[mi][ni][r] + bv));
                }
            }
        }
    }
}

// ---------------- host launcher ----------------
extern "C" void kernel_launch(void* const* d_in, const int* in_sizes, int n_in,
                              void* d_out, int out_size, void* d_ws, size_t ws_size,
                              hipStream_t stream) {
    const float* x  = (const float*)d_in[0];   // [B,S,D]
    const float* Wg = (const float*)d_in[1];   // [D,E]
    const float* W1 = (const float*)d_in[2];   // [E,D,DF]
    const float* b1 = (const float*)d_in[3];   // [E,DF]
    const float* W2 = (const float*)d_in[4];   // [E,DF,D]
    const float* b2 = (const float*)d_in[5];   // [E,D]
    float* out = (float*)d_out;                // [B,S,D] f32

    char* ws = (char*)d_ws;
    size_t off = 0;
    auto carve = [&](size_t bytes) { size_t r = off; off = (off + bytes + 255) & ~(size_t)255; return r; };
    int*            cnt   = (int*)   (ws + carve(NEXP * sizeof(int)));
    int*            basep = (int*)   (ws + carve(NEXP * sizeof(int)));
    int*            cnt2  = (int*)   (ws + carve(NEXP * sizeof(int)));
    int2*           tok_e = (int2*)  (ws + carve(TOKENS * sizeof(int2)));
    float2*         tok_p = (float2*)(ws + carve(TOKENS * sizeof(float2)));
    int*            rowsp = (int*)   (ws + carve(NASSIGN * sizeof(int)));
    float*          wtsp  = (float*) (ws + carve(NASSIGN * sizeof(float)));
    unsigned short* xb    = (unsigned short*)(ws + carve((size_t)TOKENS * DIM * 2));
    unsigned short* w1t   = (unsigned short*)(ws + carve((size_t)NEXP * DFF * DIM * 2));
    unsigned short* w2t   = (unsigned short*)(ws + carve((size_t)NEXP * DIM * DFF * 2));
    unsigned short* Hbuf  = (unsigned short*)(ws + carve((size_t)NASSIGN * DFF * 2));
    // total workspace ~290 MB

    hipMemsetAsync(out, 0, (size_t)TOKENS * DIM * sizeof(float), stream);
    hipMemsetAsync(cnt, 0, NEXP * sizeof(int), stream);

    k_convert_x<<<(TOKENS * DIM) / 1024, 256, 0, stream>>>(x, xb);
    k_transpose_bf16<<<dim3(DFF / 32, DIM / 32, NEXP), dim3(32, 8), 0, stream>>>(W1, w1t, DIM, DFF);
    k_transpose_bf16<<<dim3(DIM / 32, DFF / 32, NEXP), dim3(32, 8), 0, stream>>>(W2, w2t, DFF, DIM);

    k_gate<<<TOKENS / 8, 256, 0, stream>>>(x, Wg, cnt, tok_e, tok_p);
    k_prefix<<<1, 32, 0, stream>>>(cnt, basep, cnt2);
    k_scatter<<<TOKENS / 256, 256, 0, stream>>>(tok_e, tok_p, basep, cnt2, rowsp, wtsp);

    k_ffn1<<<dim3(DFF / TN, NASSIGN / TM, NEXP), 256, 0, stream>>>(xb, w1t, b1, rowsp, basep, cnt, Hbuf);
    k_ffn2<<<dim3(DIM / TN, NASSIGN / TM, NEXP), 256, 0, stream>>>(Hbuf, w2t, b2, rowsp, wtsp, basep, cnt, out);
}
